// Net_34892314313493
// MI455X (gfx1250) — matrix operand order repaired and checked by reading, not compiled, not run on hardware
//
#include <hip/hip_runtime.h>

// ---------------------------------------------------------------------------
// PointNet++ (PyG PointConv x3 + FPS + radius-kNN + MLP head) for gfx1250.
// GEMM work (~42 GFLOP) runs on v_wmma_f32_16x16x32_f16 with f32 accumulate.
// Working set ~20MB -> L2 resident on MI455X (192MB); matrix-core bound.
// GEMM phases keep A fragments in registers across column tiles (A reuse),
// streaming only pre-packed B fragments from L2.
// ---------------------------------------------------------------------------

typedef __attribute__((ext_vector_type(16))) _Float16 v16h;
typedef __attribute__((ext_vector_type(8)))  float    v8f;

union HFrag { v16h v; _Float16 h[16]; float4 q[2]; };

// ---- WMMA fragment helpers (layouts per CDNA5 ISA 7.12.2) -------

// A (16x32 f16): lane&15 = row M; lanes<16 hold K {0..7,16..23}, lanes>=16
// hold K {8..15,24..31}; elements 0..7 = kb..kb+7, elements 8..15 = kb+16..+23
__device__ __forceinline__ v16h load_a_frag(const _Float16* A, int row0,
                                            int lda, int k0, int lane) {
  HFrag f;
  int m  = row0 + (lane & 15);
  int kb = k0 + ((lane & 16) ? 8 : 0);
  const _Float16* p = A + (size_t)m * lda + kb;          // 16B aligned
  f.q[0] = *reinterpret_cast<const float4*>(p);          // ds_load_b128
  f.q[1] = *reinterpret_cast<const float4*>(p + 16);     // ds_load_b128
  return f.v;
}

// B (32x16 f16) pre-packed: per tile, per lane, 16 contiguous halves.
__device__ __forceinline__ v16h load_b_frag(const _Float16* W, int tile,
                                            int lane) {
  HFrag f;
  const float4* p =
      reinterpret_cast<const float4*>(W + ((size_t)tile * 32 + lane) * 16);
  f.q[0] = p[0];                                          // global_load_b128
  f.q[1] = p[1];
  return f.v;
}

__device__ __forceinline__ v8f wmma_f16(v16h a, v16h b, v8f c) {
  return __builtin_amdgcn_wmma_f32_16x16x32_f16(false, a, false, b, (short)0,
                                                c, false, false);
}

// ---- Weight packing: W[Kact x N] f32 -> zero-padded f16 B-fragment tiles ---
// tile index = nt*KT + kt (kt contiguous for the k-accumulation loop).
// B layout: col = lane&15 ; K row = kt*32 + ((lane<16)?0:8) + (h<8 ? h : h+8).
__global__ __launch_bounds__(32) void pack_weight_kernel(
    const float* __restrict__ W, int Kact, int Kpad, int N,
    _Float16* __restrict__ out) {
  int KT = Kpad >> 5;
  int tile = blockIdx.x;
  int kt = tile % KT, nt = tile / KT;
  int lane = threadIdx.x;
  int col = nt * 16 + (lane & 15);
  int kb  = kt * 32 + ((lane & 16) ? 8 : 0);
  _Float16* o = out + ((size_t)tile * 32 + lane) * 16;
#pragma unroll
  for (int h = 0; h < 16; ++h) {
    int k = kb + (h < 8 ? h : h + 8);
    float v = (k < Kact) ? W[(size_t)k * N + col] : 0.f;
    o[h] = (_Float16)v;
  }
}

// ---- radius-kNN: one wave per query point; d^2 row in LDS; 32 argmin rounds
__global__ __launch_bounds__(32) void knn_kernel(
    const float* __restrict__ pos, int n, float r2, int* __restrict__ idx_out,
    unsigned char* __restrict__ mask_out) {
  extern __shared__ float d2[];                // n floats
  __shared__ float rv[32];
  __shared__ int   ri[32];
  int b = blockIdx.x / n, i = blockIdx.x % n;
  int t = threadIdx.x;
  const float* P = pos + (size_t)b * n * 3;
  float xi = P[i * 3 + 0], yi = P[i * 3 + 1], zi = P[i * 3 + 2];
  for (int j = t; j < n; j += 32) {
    float dx = P[j * 3 + 0] - xi, dy = P[j * 3 + 1] - yi,
          dz = P[j * 3 + 2] - zi;
    float d = dx * dx + dy * dy + dz * dz;
    d2[j] = (j == i) ? 1e10f : d;              // exclude self-loop
  }
  __syncthreads();
  for (int s = 0; s < 32; ++s) {
    float best = 3e30f;
    int bi = n;
    for (int j = t; j < n; j += 32) {
      float d = d2[j];
      if (d < best) { best = d; bi = j; }      // strided: keeps lowest j
    }
    rv[t] = best; ri[t] = bi;
    __syncthreads();
    for (int off = 16; off > 0; off >>= 1) {
      if (t < off) {
        float ob = rv[t + off]; int oi = ri[t + off];
        if (ob < rv[t] || (ob == rv[t] && oi < ri[t])) { rv[t] = ob; ri[t] = oi; }
      }
      __syncthreads();
    }
    int sel = ri[0]; float selD = rv[0];
    if (t == 0) {
      idx_out[(size_t)blockIdx.x * 32 + s]  = sel;
      mask_out[(size_t)blockIdx.x * 32 + s] = (selD <= r2) ? 1 : 0;
      d2[sel] = 3e30f;                         // remove from candidates
    }
    __syncthreads();
  }
}

// ---- Farthest point sampling: one block per cloud, sequential scan --------
__global__ __launch_bounds__(256) void fps_kernel(const float* __restrict__ pos,
                                                  int n, int m,
                                                  int* __restrict__ fi) {
  extern __shared__ float mind[];              // n floats
  __shared__ float rv[256];
  __shared__ int   ri[256];
  int b = blockIdx.x, t = threadIdx.x;
  const float* P = pos + (size_t)b * n * 3;
  for (int j = t; j < n; j += 256) mind[j] = 3e38f;
  __syncthreads();
  int cur = 0;
  for (int s = 0; s < m; ++s) {
    if (t == 0) fi[(size_t)b * m + s] = cur;   // scan emits cur, then updates
    float cx = P[cur * 3 + 0], cy = P[cur * 3 + 1], cz = P[cur * 3 + 2];
    float bv = -1.f; int bi = 0;
    for (int j = t; j < n; j += 256) {
      float dx = P[j * 3 + 0] - cx, dy = P[j * 3 + 1] - cy,
            dz = P[j * 3 + 2] - cz;
      float d = dx * dx + dy * dy + dz * dz;
      float mv = fminf(mind[j], d);
      mind[j] = mv;
      if (mv > bv || (mv == bv && j < bi)) { bv = mv; bi = j; }
    }
    rv[t] = bv; ri[t] = bi;
    __syncthreads();
    for (int off = 128; off > 0; off >>= 1) {
      if (t < off) {
        float ob = rv[t + off]; int oi = ri[t + off];
        if (ob > rv[t] || (ob == rv[t] && oi < ri[t])) { rv[t] = ob; ri[t] = oi; }
      }
      __syncthreads();
    }
    cur = ri[0];
    __syncthreads();
  }
}

// ---- Gather x[b, fi[b,j], c] -> out[b, j, c] ------------------------------
__global__ __launch_bounds__(256) void gather_kernel(
    const float* __restrict__ in, const int* __restrict__ fi,
    float* __restrict__ out, int n, int m, int C, long total) {
  long i = (long)blockIdx.x * 256 + threadIdx.x;
  if (i >= total) return;
  int c = (int)(i % C);
  long rest = i / C;
  int j = (int)(rest % m);
  int b = (int)(rest / m);
  out[i] = in[((size_t)b * n + fi[(size_t)b * m + j]) * C + c];
}

// ---- Fused PointConv: gather -> WMMA GEMM1 -> ReLU -> WMMA GEMM2 ----------
//      -> masked neighbor max (bitwise uint atomicMax; values clamped >= 0).
// One workgroup = 2 points x 32 neighbors = 64-row M tile, 8 waves.
// Wave mapping: wave = rowtile*2 + half; A fragments for the row tile are
// loaded once into registers and reused across all column tiles.
template <int CX, int CIN_PAD, int H>
__global__ __launch_bounds__(256) void pointconv_kernel(
    const float* __restrict__ x,    // [B,n,CX] or nullptr when CX==0
    const float* __restrict__ pos,  // [B,n,3]
    const int* __restrict__ idx,    // [B,n,32]
    const unsigned char* __restrict__ mask,
    const _Float16* __restrict__ w1p, const float* __restrict__ b1,
    const _Float16* __restrict__ w2p, const float* __restrict__ b2,
    float* __restrict__ outx,       // [B,n,H]
    int n) {
  constexpr int ROWS = 64;
  __shared__ __align__(16) _Float16 Abuf[ROWS * CIN_PAD];
  __shared__ __align__(16) _Float16 Hbuf[ROWS * H];
  __shared__ unsigned int maxbits[2 * H];
  __shared__ unsigned char mrow[ROWS];

  int nb = n >> 1;
  int b = blockIdx.x / nb;
  int p0 = (blockIdx.x % nb) * 2;
  int tid = threadIdx.x;
  for (int i = tid; i < 2 * H; i += 256) maxbits[i] = 0u;  // bits of +0.0f

  // ---- build A = [x_j, pos_j - pos_i] (f16, zero-padded to CIN_PAD) ----
  const float* P = pos + (size_t)b * n * 3;
  {
    int r = tid >> 2, sub = tid & 3;            // 4 threads per row
    int p = p0 + (r >> 5), j = r & 31;
    size_t base = ((size_t)b * n + p) * 32 + j;
    int g = idx[base];
    if (sub == 0) mrow[r] = mask[base];
    float rx = P[g * 3 + 0] - P[p * 3 + 0];
    float ry = P[g * 3 + 1] - P[p * 3 + 1];
    float rz = P[g * 3 + 2] - P[p * 3 + 2];
    const float* Xg = (CX > 0) ? (x + ((size_t)b * n + g) * CX) : nullptr;
    _Float16* Ar = Abuf + r * CIN_PAD;
    for (int c = sub; c < CIN_PAD; c += 4) {
      float v;
      if (c < CX)            v = Xg[c];
      else if (c == CX)      v = rx;
      else if (c == CX + 1)  v = ry;
      else if (c == CX + 2)  v = rz;
      else                   v = 0.f;
      Ar[c] = (_Float16)v;
    }
  }
  __syncthreads();

  int wave = tid >> 5, lane = tid & 31;
  constexpr int CT  = H / 16;      // col tiles (square hidden layers)
  constexpr int KT1 = CIN_PAD / 32;
  constexpr int KT2 = H / 32;
  int tr   = (wave >> 1) << 4;     // this wave's 16-row tile (0,16,32,48)
  int half = wave & 1;             // column-tile split between wave pairs
  int rbase = tr + ((lane >> 4) << 3);

  // ---- GEMM1: Hbuf = relu(A @ W1 + b1), f16 in LDS ----
  {
    v16h afr[KT1];
#pragma unroll
    for (int kt = 0; kt < KT1; ++kt)
      afr[kt] = load_a_frag(Abuf, tr, CIN_PAD, kt * 32, lane);
    for (int tc = half; tc < CT; tc += 2) {
      v8f acc = {};
#pragma unroll
      for (int kt = 0; kt < KT1; ++kt)
        acc = wmma_f16(afr[kt], load_b_frag(w1p, tc * KT1 + kt, lane), acc);
      int col = tc * 16 + (lane & 15);
      float bias = b1[col];
#pragma unroll
      for (int i = 0; i < 8; ++i) {
        float v = fmaxf(acc[i] + bias, 0.f);
        Hbuf[(rbase + i) * H + col] = (_Float16)v;
      }
    }
  }
  __syncthreads();

  // ---- GEMM2 + masked neighbor max: relu(where(any, max_j h2, 0)) ----
  //      == max(0, h2 over valid j)  -> monotone in uint bit pattern.
  {
    v16h afr[KT2];
#pragma unroll
    for (int kt = 0; kt < KT2; ++kt)
      afr[kt] = load_a_frag(Hbuf, tr, H, kt * 32, lane);
    int pl = tr >> 5;                         // local point of this row tile
    for (int tc = half; tc < CT; tc += 2) {
      v8f acc = {};
#pragma unroll
      for (int kt = 0; kt < KT2; ++kt)
        acc = wmma_f16(afr[kt], load_b_frag(w2p, tc * KT2 + kt, lane), acc);
      int col = tc * 16 + (lane & 15);
      float bias = b2[col];
#pragma unroll
      for (int i = 0; i < 8; ++i) {
        int rr = rbase + i;
        if (mrow[rr]) {
          unsigned int bits = __float_as_uint(fmaxf(acc[i] + bias, 0.f));
          atomicMax(&maxbits[pl * H + col], bits);
        }
      }
    }
  }
  __syncthreads();

  for (int i = tid; i < 2 * H; i += 256) {
    int pl = i / H, col = i % H;
    outx[((size_t)b * n + p0 + pl) * H + col] = __uint_as_float(maxbits[i]);
  }
}

// ---- Head: global max pool + Lin/ReLU x2 + Lin + log_softmax --------------
__global__ __launch_bounds__(256) void head_kernel(
    const float* __restrict__ x3,  // [B,npts,256]
    const float* __restrict__ wl1, const float* __restrict__ bl1,
    const float* __restrict__ wl2, const float* __restrict__ bl2,
    const float* __restrict__ wl3, const float* __restrict__ bl3,
    float* __restrict__ out, int npts) {
  __shared__ float pooled[256], y1[256], y2[256], logits[16];
  int b = blockIdx.x, c = threadIdx.x;
  const float* X = x3 + (size_t)b * npts * 256;
  float mx = -3e38f;
  for (int p = 0; p < npts; ++p) mx = fmaxf(mx, X[(size_t)p * 256 + c]);
  pooled[c] = mx;
  __syncthreads();
  float acc = bl1[c];
  for (int k = 0; k < 256; ++k) acc += pooled[k] * wl1[(size_t)k * 256 + c];
  y1[c] = fmaxf(acc, 0.f);
  __syncthreads();
  acc = bl2[c];
  for (int k = 0; k < 256; ++k) acc += y1[k] * wl2[(size_t)k * 256 + c];
  y2[c] = fmaxf(acc, 0.f);
  __syncthreads();
  if (c < 10) {
    acc = bl3[c];
    for (int k = 0; k < 256; ++k) acc += y2[k] * wl3[(size_t)k * 10 + c];
    logits[c] = acc;
  }
  __syncthreads();
  if (c == 0) {
    float m = -3e38f;
    for (int i = 0; i < 10; ++i) m = fmaxf(m, logits[i]);
    float s = 0.f;
    for (int i = 0; i < 10; ++i) s += expf(logits[i] - m);
    float lse = m + logf(s);
    for (int i = 0; i < 10; ++i) out[b * 10 + i] = logits[i] - lse;
  }
}

// ---------------------------------------------------------------------------
extern "C" void kernel_launch(void* const* d_in, const int* in_sizes, int n_in,
                              void* d_out, int out_size, void* d_ws,
                              size_t ws_size, hipStream_t stream) {
  const int B = 8, N0 = 2048, N1 = 1024, N2 = 512;
  const float* pos = (const float*)d_in[0];
  // d_in[1] = batch (implicit in dense layout, unused)
  const float* w1a = (const float*)d_in[2];  const float* b1a = (const float*)d_in[3];
  const float* w1b = (const float*)d_in[4];  const float* b1b = (const float*)d_in[5];
  const float* w2a = (const float*)d_in[6];  const float* b2a = (const float*)d_in[7];
  const float* w2b = (const float*)d_in[8];  const float* b2b = (const float*)d_in[9];
  const float* w3a = (const float*)d_in[10]; const float* b3a = (const float*)d_in[11];
  const float* w3b = (const float*)d_in[12]; const float* b3b = (const float*)d_in[13];
  const float* wl1 = (const float*)d_in[14]; const float* bl1 = (const float*)d_in[15];
  const float* wl2 = (const float*)d_in[16]; const float* bl2 = (const float*)d_in[17];
  const float* wl3 = (const float*)d_in[18]; const float* bl3 = (const float*)d_in[19];

  char* ws = (char*)d_ws;
  size_t off = 0;
  auto alloc = [&](size_t bytes) -> void* {
    off = (off + 255) & ~(size_t)255;
    void* p = ws + off;
    off += bytes;
    return p;
  };

  _Float16* w1aP = (_Float16*)alloc((size_t)32 * 64 * 2);
  _Float16* w1bP = (_Float16*)alloc((size_t)64 * 64 * 2);
  _Float16* w2aP = (_Float16*)alloc((size_t)96 * 128 * 2);
  _Float16* w2bP = (_Float16*)alloc((size_t)128 * 128 * 2);
  _Float16* w3aP = (_Float16*)alloc((size_t)160 * 256 * 2);
  _Float16* w3bP = (_Float16*)alloc((size_t)256 * 256 * 2);
  int*           idx1  = (int*)alloc((size_t)B * N0 * 32 * 4);
  unsigned char* mask1 = (unsigned char*)alloc((size_t)B * N0 * 32);
  float*         x1    = (float*)alloc((size_t)B * N0 * 64 * 4);
  int*           fi1   = (int*)alloc((size_t)B * N1 * 4);
  float*         pos1  = (float*)alloc((size_t)B * N1 * 3 * 4);
  float*         x1g   = (float*)alloc((size_t)B * N1 * 64 * 4);
  int*           idx2  = (int*)alloc((size_t)B * N1 * 32 * 4);
  unsigned char* mask2 = (unsigned char*)alloc((size_t)B * N1 * 32);
  float*         x2    = (float*)alloc((size_t)B * N1 * 128 * 4);
  int*           fi2   = (int*)alloc((size_t)B * N2 * 4);
  float*         pos2  = (float*)alloc((size_t)B * N2 * 3 * 4);
  float*         x2g   = (float*)alloc((size_t)B * N2 * 128 * 4);
  int*           idx3  = (int*)alloc((size_t)B * N2 * 32 * 4);
  unsigned char* mask3 = (unsigned char*)alloc((size_t)B * N2 * 32);
  float*         x3    = (float*)alloc((size_t)B * N2 * 256 * 4);
  (void)ws_size; (void)in_sizes; (void)n_in; (void)out_size;

  // Pack weights into WMMA B-fragment layout (K zero-padded to mult. of 32).
  pack_weight_kernel<<<(32 / 32) * (64 / 16), 32, 0, stream>>>(w1a, 3, 32, 64, w1aP);
  pack_weight_kernel<<<(64 / 32) * (64 / 16), 32, 0, stream>>>(w1b, 64, 64, 64, w1bP);
  pack_weight_kernel<<<(96 / 32) * (128 / 16), 32, 0, stream>>>(w2a, 67, 96, 128, w2aP);
  pack_weight_kernel<<<(128 / 32) * (128 / 16), 32, 0, stream>>>(w2b, 128, 128, 128, w2bP);
  pack_weight_kernel<<<(160 / 32) * (256 / 16), 32, 0, stream>>>(w3a, 131, 160, 256, w3aP);
  pack_weight_kernel<<<(256 / 32) * (256 / 16), 32, 0, stream>>>(w3b, 256, 256, 256, w3bP);

  // Stage 1: r=0.2 kNN, PointConv(3 -> 64 -> 64)
  knn_kernel<<<B * N0, 32, N0 * 4, stream>>>(pos, N0, 0.04f, idx1, mask1);
  pointconv_kernel<0, 32, 64><<<B * (N0 / 2), 256, 0, stream>>>(
      nullptr, pos, idx1, mask1, w1aP, b1a, w1bP, b1b, x1, N0);
  // FPS ratio 0.5 + gather
  fps_kernel<<<B, 256, N0 * 4, stream>>>(pos, N0, N1, fi1);
  {
    long tot = (long)B * N1 * 64;
    gather_kernel<<<(int)((tot + 255) / 256), 256, 0, stream>>>(x1, fi1, x1g, N0, N1, 64, tot);
    tot = (long)B * N1 * 3;
    gather_kernel<<<(int)((tot + 255) / 256), 256, 0, stream>>>(pos, fi1, pos1, N0, N1, 3, tot);
  }

  // Stage 2: r=0.4 kNN, PointConv(67 -> 128 -> 128)
  knn_kernel<<<B * N1, 32, N1 * 4, stream>>>(pos1, N1, 0.16f, idx2, mask2);
  pointconv_kernel<64, 96, 128><<<B * (N1 / 2), 256, 0, stream>>>(
      x1g, pos1, idx2, mask2, w2aP, b2a, w2bP, b2b, x2, N1);
  // FPS ratio 0.5 + gather
  fps_kernel<<<B, 256, N1 * 4, stream>>>(pos1, N1, N2, fi2);
  {
    long tot = (long)B * N2 * 128;
    gather_kernel<<<(int)((tot + 255) / 256), 256, 0, stream>>>(x2, fi2, x2g, N1, N2, 128, tot);
    tot = (long)B * N2 * 3;
    gather_kernel<<<(int)((tot + 255) / 256), 256, 0, stream>>>(pos1, fi2, pos2, N1, N2, 3, tot);
  }

  // Stage 3: r=1.0 kNN, PointConv(131 -> 256 -> 256)
  knn_kernel<<<B * N2, 32, N2 * 4, stream>>>(pos2, N2, 1.0f, idx3, mask3);
  pointconv_kernel<128, 160, 256><<<B * (N2 / 2), 256, 0, stream>>>(
      x2g, pos2, idx3, mask3, w3aP, b3a, w3bP, b3b, x3, N2);

  // Head: global max pool + MLP + log_softmax -> d_out [B,10] f32
  head_kernel<<<B, 256, 0, stream>>>(x3, wl1, bl1, wl2, bl2, wl3, bl3,
                                     (float*)d_out, N2);
}
